// RNN_47347719471533
// MI455X (gfx1250) — compile-verified, matrix operand order checked by reading, not run on hardware
//
#include <hip/hip_runtime.h>
#include <hip/hip_bf16.h>
#include <math.h>

// Sizes from the reference: B=32, S=64, V=32000, E=300, H=1024
#define HDIM   1024
#define BATCH  32
#define SEQ    64
#define NSTEP  63          // S-1
#define VOCAB  32000
#define E_RAW  300
#define E_PAD  320         // padded to multiple of 32 for WMMA K

typedef __attribute__((ext_vector_type(16))) __bf16 v16bf;
typedef __attribute__((ext_vector_type(8)))  float  v8f;

// ---------------------------------------------------------------------------
// WMMA fragment helpers (v_wmma_f32_16x16x32_bf16, wave32)
// 16-bit A-matrix 16x32 layout (ISA 7.12.2): lane 0-15 -> row=lane, holds
// K = k0..k0+7 in VGPR0-3 and K = k0+16..k0+23 in VGPR4-7; lanes 16-31 hold
// the other K halves. B (N x K row-major weights) uses the identical mapping
// with N in place of M. Both are two contiguous 16-byte loads per lane.
// ---------------------------------------------------------------------------
__device__ __forceinline__ v16bf load_frag_bf16(const __bf16* base, int ld,
                                                int row0, int k0) {
  int lane = threadIdx.x & 31;
  int half = lane >> 4;
  int r    = lane & 15;
  const __bf16* p = base + (size_t)(row0 + r) * ld + k0 + half * 8;
  union { v16bf v; uint4 q[2]; } u;
  u.q[0] = *(const uint4*)(p);        // K pairs 0..7   (of this 32-slice half)
  u.q[1] = *(const uint4*)(p + 16);   // K pairs 16..23
  return u.v;
}

__device__ __forceinline__ v8f wmma_bf16(v16bf a, v16bf b, v8f c) {
  return __builtin_amdgcn_wmma_f32_16x16x32_bf16(
      /*neg_a=*/false, a, /*neg_b=*/false, b,
      /*c_mod=*/(short)0, c, /*reuse_a=*/false, /*reuse_b=*/false);
}

// ---------------------------------------------------------------------------
// Utility kernels
// ---------------------------------------------------------------------------
__global__ void zero_f32_kernel(float* p, int n) {
  int i = blockIdx.x * blockDim.x + threadIdx.x;
  if (i < n) p[i] = 0.0f;
}

__global__ void zero_bf16_kernel(__bf16* p, int n) {
  int i = blockIdx.x * blockDim.x + threadIdx.x;
  if (i < n) p[i] = (__bf16)0.0f;
}

// fp32 -> bf16 row-wise convert with zero padding of extra columns
__global__ void cvt_bf16_kernel(const float* __restrict__ src,
                                __bf16* __restrict__ dst,
                                int scols, int dcols) {
  int r = blockIdx.x;
  const float* s = src + (size_t)r * scols;
  __bf16* d = dst + (size_t)r * dcols;
  for (int c = threadIdx.x; c < dcols; c += blockDim.x)
    d[c] = (c < scols) ? (__bf16)s[c] : (__bf16)0.0f;
}

// x[t][b][:] = bf16(emb[sent[b][t]][:]) padded E_RAW -> E_PAD
__global__ void embed_kernel(const int* __restrict__ sent,
                             const float* __restrict__ emb,
                             __bf16* __restrict__ xb) {
  int r = blockIdx.x;              // r = t*32 + b, t in [0, NSTEP)
  int t = r >> 5;
  int b = r & 31;
  int tok = sent[b * SEQ + t];
  const float* e = emb + (size_t)tok * E_RAW;
  __bf16* x = xb + (size_t)r * E_PAD;
  for (int c = threadIdx.x; c < E_PAD; c += blockDim.x)
    x[c] = (c < E_RAW) ? (__bf16)e[c] : (__bf16)0.0f;
}

// ---------------------------------------------------------------------------
// One GRU layer step: h_new = GRU(x, h_prev)
//   gates = [x | h] @ [Wi | Wh].T ; r,z combined-accumulated, n kept split
//   (n = tanh(nx + bi_n + r * (nh + bh_n)))
// Grid: 32 blocks x 128 threads = 128 waves; wave = one (16-batch,16-hidden)
// tile; 4 f32 accumulators per wave.
// ---------------------------------------------------------------------------
__global__ void gru_layer_kernel(const __bf16* __restrict__ X, int Kx,   // 32 x Kx
                                 const __bf16* __restrict__ Hb,         // 32 x H (bf16 prev)
                                 const float*  __restrict__ Hf,         // 32 x H (f32 prev)
                                 const __bf16* __restrict__ Wi,         // 3H x Kx
                                 const __bf16* __restrict__ Wh,         // 3H x H
                                 const float*  __restrict__ bi,
                                 const float*  __restrict__ bh,
                                 float*  __restrict__ Hf_new,           // 32 x H
                                 __bf16* __restrict__ Hb_new) {         // 32 x H
  int tile = blockIdx.x * 4 + (threadIdx.x >> 5);  // 0..127
  int mt = tile & 1;          // batch tile (0..1)
  int it = tile >> 1;         // hidden tile (0..63)
  int b0 = mt << 4;
  int i0 = it << 4;

  v8f ar = {}, az = {}, anx = {}, anh = {};

  // x contribution (K = Kx)
  for (int k0 = 0; k0 < Kx; k0 += 32) {
    v16bf a = load_frag_bf16(X, Kx, b0, k0);
    ar  = wmma_bf16(a, load_frag_bf16(Wi, Kx, i0,            k0), ar);
    az  = wmma_bf16(a, load_frag_bf16(Wi, Kx, i0 + HDIM,     k0), az);
    anx = wmma_bf16(a, load_frag_bf16(Wi, Kx, i0 + 2 * HDIM, k0), anx);
  }
  // h contribution (K = H)
  for (int k0 = 0; k0 < HDIM; k0 += 32) {
    v16bf a = load_frag_bf16(Hb, HDIM, b0, k0);
    ar  = wmma_bf16(a, load_frag_bf16(Wh, HDIM, i0,            k0), ar);
    az  = wmma_bf16(a, load_frag_bf16(Wh, HDIM, i0 + HDIM,     k0), az);
    anh = wmma_bf16(a, load_frag_bf16(Wh, HDIM, i0 + 2 * HDIM, k0), anh);
  }

  // elementwise gate math; C/D layout: VGPR d -> M = d (+8 for upper lanes),
  // N = lane & 15
  int lane = threadIdx.x & 31;
  int n  = lane & 15;
  int dm = (lane >> 4) << 3;
  int i  = i0 + n;
  float br  = bi[i]            + bh[i];
  float bz  = bi[i + HDIM]     + bh[i + HDIM];
  float bnx = bi[i + 2 * HDIM];
  float bnh = bh[i + 2 * HDIM];
#pragma unroll
  for (int d = 0; d < 8; d++) {
    int b = b0 + d + dm;
    float r  = 1.0f / (1.0f + __expf(-(ar[d] + br)));
    float z  = 1.0f / (1.0f + __expf(-(az[d] + bz)));
    float nv = tanhf(anx[d] + bnx + r * (anh[d] + bnh));
    float hp = Hf[b * HDIM + i];
    float hn = (1.0f - z) * nv + z * hp;
    Hf_new[b * HDIM + i] = hn;
    Hb_new[b * HDIM + i] = (__bf16)hn;
  }
}

// ---------------------------------------------------------------------------
// Output projection: logits[b,t,v] = sum_h hs[t*32+b, h] * Wout[v, h] + bout[v]
// M = 2016 (126 tiles), N = 32000, K = 1024. Each wave: 16x64 output
// (4 accumulators), A fragment reused across the 4 N-tiles.
// Grid: 7875 blocks x 256 threads (63000 wave-tasks, exact).
// ---------------------------------------------------------------------------
__global__ void out_gemm_kernel(const __bf16* __restrict__ A,   // 2016 x 1024
                                const __bf16* __restrict__ Wb,  // 32000 x 1024
                                const float*  __restrict__ bo,  // 32000
                                float* __restrict__ out) {      // d_out: [loss | logits]
  int wave = threadIdx.x >> 5;
  int task = blockIdx.x * 8 + wave;        // 0..62999
  int mt = task % 126;
  int ng = task / 126;
  int r0 = mt << 4;
  int n0 = ng << 6;

  v8f acc0 = {}, acc1 = {}, acc2 = {}, acc3 = {};
  for (int k0 = 0; k0 < HDIM; k0 += 32) {
    v16bf a = load_frag_bf16(A, HDIM, r0, k0);
    acc0 = wmma_bf16(a, load_frag_bf16(Wb, HDIM, n0,      k0), acc0);
    acc1 = wmma_bf16(a, load_frag_bf16(Wb, HDIM, n0 + 16, k0), acc1);
    acc2 = wmma_bf16(a, load_frag_bf16(Wb, HDIM, n0 + 32, k0), acc2);
    acc3 = wmma_bf16(a, load_frag_bf16(Wb, HDIM, n0 + 48, k0), acc3);
  }

  int lane = threadIdx.x & 31;
  int nn = lane & 15;
  int dm = (lane >> 4) << 3;
  v8f accs[4] = {acc0, acc1, acc2, acc3};
#pragma unroll
  for (int j = 0; j < 4; j++) {
    int v = n0 + j * 16 + nn;
    float bv = bo[v];
#pragma unroll
    for (int d = 0; d < 8; d++) {
      int R  = r0 + d + dm;       // R = t*32 + b
      int bb = R & 31;
      int tt = R >> 5;
      out[1 + (size_t)(bb * NSTEP + tt) * VOCAB + v] = accs[j][d] + bv;
    }
  }
}

// ---------------------------------------------------------------------------
// Per-(b,t) log-softmax + target NLL, masked, averaged. One block per row.
// ---------------------------------------------------------------------------
__global__ void loss_kernel(float* __restrict__ out,
                            const int* __restrict__ sent,
                            const int* __restrict__ length) {
  int r = blockIdx.x;          // r = b*63 + t (matches logits row layout)
  int b = r / NSTEP;
  int t = r % NSTEP;
  const float* row = out + 1 + (size_t)r * VOCAB;
  int tid = threadIdx.x;
  __shared__ float sred[8];

  float m = -3.4e38f;
  for (int v = tid; v < VOCAB; v += 256) m = fmaxf(m, row[v]);
#pragma unroll
  for (int off = 16; off > 0; off >>= 1) m = fmaxf(m, __shfl_xor(m, off, 32));
  if ((tid & 31) == 0) sred[tid >> 5] = m;
  __syncthreads();
  float mm = sred[0];
#pragma unroll
  for (int i = 1; i < 8; i++) mm = fmaxf(mm, sred[i]);
  __syncthreads();

  float s = 0.0f;
  for (int v = tid; v < VOCAB; v += 256) s += __expf(row[v] - mm);
#pragma unroll
  for (int off = 16; off > 0; off >>= 1) s += __shfl_xor(s, off, 32);
  if ((tid & 31) == 0) sred[tid >> 5] = s;
  __syncthreads();

  if (tid == 0) {
    float stot = 0.0f;
#pragma unroll
    for (int i = 0; i < 8; i++) stot += sred[i];
    int tgt = sent[b * SEQ + t + 1];
    int len = length[b] - 1;
    if (t < len) {
      float lp = row[tgt] - mm - __logf(stot);
      atomicAdd(out, -lp / ((float)len * 32.0f));
    }
  }
}

// ---------------------------------------------------------------------------
// Host launcher
// ---------------------------------------------------------------------------
extern "C" void kernel_launch(void* const* d_in, const int* in_sizes, int n_in,
                              void* d_out, int out_size, void* d_ws, size_t ws_size,
                              hipStream_t stream) {
  (void)in_sizes; (void)n_in; (void)out_size; (void)ws_size;

  const int*   sent   = (const int*)d_in[0];
  const int*   length = (const int*)d_in[1];
  const float* emb    = (const float*)d_in[2];
  const float* Wi0    = (const float*)d_in[3];
  const float* Wh0    = (const float*)d_in[4];
  const float* bi0    = (const float*)d_in[5];
  const float* bh0    = (const float*)d_in[6];
  const float* Wi1    = (const float*)d_in[7];
  const float* Wh1    = (const float*)d_in[8];
  const float* bi1    = (const float*)d_in[9];
  const float* bh1    = (const float*)d_in[10];
  const float* Wout   = (const float*)d_in[11];
  const float* boutp  = (const float*)d_in[12];
  float* out = (float*)d_out;

  // workspace carve-up (all offsets 256B aligned; hipMalloc base is aligned)
  char* ws = (char*)d_ws;
  size_t off = 0;
  auto alloc = [&](size_t bytes) -> void* {
    void* p = ws + off;
    off += (bytes + 255) & ~(size_t)255;
    return p;
  };
  __bf16* Wi0b  = (__bf16*)alloc((size_t)3 * HDIM * E_PAD * 2);        // 3072 x 320
  __bf16* Wh0b  = (__bf16*)alloc((size_t)3 * HDIM * HDIM * 2);         // 3072 x 1024
  __bf16* Wi1b  = (__bf16*)alloc((size_t)3 * HDIM * HDIM * 2);
  __bf16* Wh1b  = (__bf16*)alloc((size_t)3 * HDIM * HDIM * 2);
  __bf16* Woutb = (__bf16*)alloc((size_t)VOCAB * HDIM * 2);            // 32000 x 1024
  __bf16* xb    = (__bf16*)alloc((size_t)NSTEP * BATCH * E_PAD * 2);   // 63 x 32 x 320
  __bf16* hs    = (__bf16*)alloc((size_t)(NSTEP + 1) * BATCH * HDIM * 2); // block0 = h1 init
  __bf16* h0b   = (__bf16*)alloc((size_t)2 * BATCH * HDIM * 2);        // ping-pong
  float*  h0f   = (float*) alloc((size_t)2 * BATCH * HDIM * 4);
  float*  h1f   = (float*) alloc((size_t)2 * BATCH * HDIM * 4);

  const int BH = BATCH * HDIM;  // 32768

  // weight / activation precision conversion (fp32 -> bf16, K padded)
  cvt_bf16_kernel<<<3 * HDIM, 256, 0, stream>>>(Wi0, Wi0b, E_RAW, E_PAD);
  cvt_bf16_kernel<<<3 * HDIM, 256, 0, stream>>>(Wh0, Wh0b, HDIM, HDIM);
  cvt_bf16_kernel<<<3 * HDIM, 256, 0, stream>>>(Wi1, Wi1b, HDIM, HDIM);
  cvt_bf16_kernel<<<3 * HDIM, 256, 0, stream>>>(Wh1, Wh1b, HDIM, HDIM);
  cvt_bf16_kernel<<<VOCAB, 256, 0, stream>>>(Wout, Woutb, HDIM, HDIM);
  embed_kernel<<<NSTEP * BATCH, 256, 0, stream>>>(sent, emb, xb);

  // zero initial states + loss accumulator
  zero_bf16_kernel<<<(BH + 255) / 256, 256, 0, stream>>>(h0b, BH);     // h0 buf0
  zero_bf16_kernel<<<(BH + 255) / 256, 256, 0, stream>>>(hs, BH);      // h1 init block
  zero_f32_kernel<<<(BH + 255) / 256, 256, 0, stream>>>(h0f, BH);
  zero_f32_kernel<<<(BH + 255) / 256, 256, 0, stream>>>(h1f, BH);
  zero_f32_kernel<<<1, 1, 0, stream>>>(out, 1);

  // sequential 2-layer GRU recurrence, 63 steps
  for (int t = 0; t < NSTEP; t++) {
    int pi = t & 1, po = pi ^ 1;
    // layer 0: x_t (32 x 320) , h0
    gru_layer_kernel<<<32, 128, 0, stream>>>(
        xb + (size_t)t * BATCH * E_PAD, E_PAD,
        h0b + (size_t)pi * BH, h0f + (size_t)pi * BH,
        Wi0b, Wh0b, bi0, bh0,
        h0f + (size_t)po * BH, h0b + (size_t)po * BH);
    // layer 1: input = new h0 (32 x 1024), h1 ping-pongs through hs blocks
    gru_layer_kernel<<<32, 128, 0, stream>>>(
        h0b + (size_t)po * BH, HDIM,
        hs + (size_t)t * BH, h1f + (size_t)pi * BH,
        Wi1b, Wh1b, bi1, bh1,
        h1f + (size_t)po * BH, hs + (size_t)(t + 1) * BH);
  }

  // big output GEMM: (2016 x 1024) x (1024 x 32000) -> logits in d_out[1..]
  out_gemm_kernel<<<7875, 256, 0, stream>>>(hs + (size_t)BH, Woutb, boutp, out);

  // masked NLL loss -> d_out[0]
  loss_kernel<<<BATCH * NSTEP, 256, 0, stream>>>(out, sent, length);
}
